// RecurrentFlowUnit_62388694942283
// MI455X (gfx1250) — matrix operand order repaired and checked by reading, not compiled, not run on hardware
//
#include <hip/hip_runtime.h>
#include <hip/hip_bf16.h>

typedef __attribute__((ext_vector_type(16))) _Float16 v16h;
typedef __attribute__((ext_vector_type(8)))  _Float16 v8h;
typedef __attribute__((ext_vector_type(8)))  float    v8f;

#define NIMG 81
#define IMGH 64
#define IMGW 96

// ---------------------------------------------------------------------------
// Warp + concat: build X0 = [81*64*96, 64] fp16 NHWC
// ---------------------------------------------------------------------------
__global__ void warp_concat_kernel(const float* __restrict__ feat1,
                                   const float* __restrict__ feat2,
                                   const float* __restrict__ coords,
                                   _Float16* __restrict__ x0)
{
    const int NPIX = IMGH * IMGW;
    const int idx = blockIdx.x * blockDim.x + threadIdx.x;
    if (idx >= NIMG * NPIX) return;
    const int img = idx / NPIX;
    const int pix = idx % NPIX;
    const int v = img / 9, u = img % 9;
    const float cx = coords[pix];
    const float cy = coords[NPIX + pix];
    const float xx = cx + (float)(u - 4);
    const float yy = cy + (float)(v - 4);
    const float x0f = floorf(xx), y0f = floorf(yy);
    const float wx1 = xx - x0f, wx0 = 1.f - wx1;
    const float wy1 = yy - y0f, wy0 = 1.f - wy1;
    const int xi = (int)x0f, yi = (int)y0f;

    int   cidx[4];
    float cw[4];
#pragma unroll
    for (int k = 0; k < 4; ++k) {
        const int xk = xi + (k & 1);
        const int yk = yi + (k >> 1);
        const bool ok = (xk >= 0) && (xk < IMGW) && (yk >= 0) && (yk < IMGH);
        const int xc = min(max(xk, 0), IMGW - 1);
        const int yc = min(max(yk, 0), IMGH - 1);
        cidx[k] = yc * IMGW + xc;
        cw[k] = ((k & 1) ? wx1 : wx0) * ((k >> 1) ? wy1 : wy0) * (ok ? 1.f : 0.f);
    }
    _Float16* dst = x0 + (long)idx * 64;
#pragma unroll
    for (int c = 0; c < 32; ++c) {
        dst[c] = (_Float16)feat1[c * NPIX + pix];
        const float* f2c = feat2 + c * NPIX;
        const float s = cw[0] * f2c[cidx[0]] + cw[1] * f2c[cidx[1]]
                      + cw[2] * f2c[cidx[2]] + cw[3] * f2c[cidx[3]];
        dst[32 + c] = (_Float16)s;
    }
}

// ---------------------------------------------------------------------------
// Weight packing into per-lane WMMA B fragments in contiguous k order (same as A) (32K x 16N per tile):
// lane<16 holds K 0..15, lane>=16 holds K 16..31, N = lane%16.
// packed[chunk][colTile][lane][16]
// ---------------------------------------------------------------------------
__global__ void pack_conv_w(const float* __restrict__ w, _Float16* __restrict__ out,
                            int CIN, int COUT, int KH, int KW)
{
    const int nColTiles = COUT / 16;
    const int cpt = CIN / 32;
    const int nChunks = KH * KW * cpt;
    const int total = nChunks * nColTiles * 512;
    const int idx = blockIdx.x * blockDim.x + threadIdx.x;
    if (idx >= total) return;
    const int h    = idx & 15;
    const int lane = (idx >> 4) & 31;
    const int ctc  = idx >> 9;
    const int ct   = ctc % nColTiles;
    const int ch   = ctc / nColTiles;
    const int t    = ch / cpt;
    const int ci0  = (ch % cpt) * 32;
    const int K    = (lane < 16 ? 0 : 16) + h;
    const int n    = lane & 15;
    const int ci   = ci0 + K;
    const int co   = ct * 16 + n;
    const int ky = t / KW, kx = t % KW;
    out[idx] = (_Float16)w[(((long)co * CIN + ci) * KH + ky) * KW + kx];
}

// Deconv (ConvTranspose2d k=4 s=2 p=1) split by output parity (py,px).
// w5t layout (I=64, O=32, 4, 4)
__global__ void pack_deconv_w(const float* __restrict__ w5t, _Float16* __restrict__ out,
                              int py, int px)
{
    const int total = 8 * 2 * 512;
    const int idx = blockIdx.x * blockDim.x + threadIdx.x;
    if (idx >= total) return;
    const int h    = idx & 15;
    const int lane = (idx >> 4) & 31;
    const int ctc  = idx >> 9;
    const int ct   = ctc % 2;
    const int ch   = ctc / 2;
    const int t    = ch / 2;
    const int ci0  = (ch % 2) * 32;
    const int K    = (lane < 16 ? 0 : 16) + h;
    const int n    = lane & 15;
    const int ci   = ci0 + K;
    const int co   = ct * 16 + n;
    const int kyI = t >> 1, kxI = t & 1;
    const int ky = (1 - py) + 2 * kyI;
    const int kx = (1 - px) + 2 * kxI;
    out[idx] = (_Float16)w5t[(((long)ci * 32 + co) * 4 + ky) * 4 + kx];
}

// ---------------------------------------------------------------------------
// Implicit-GEMM conv on v_wmma_f32_16x16x32_f16.
// 256 threads = 8 waves (4M x 2N); tile = 128 pixels x COUT.
// A (im2col) streamed Global->LDS with global_load_async_to_lds_b128 into a
// double-buffered tile (ASYNCcnt); B read pre-swizzled from global into regs.
// Inputs are pre-normalized (BN+ReLU applied by bn_apply), so staging is a
// pure copy. Epilogue: raw fp16 NHWC out + per-channel sum/sumsq for BN.
// ---------------------------------------------------------------------------
template<int CIN, int COUT, int S, int NTAPS, bool DECONV>
__global__ __launch_bounds__(256)
void conv_igemm(const _Float16* __restrict__ actIn,
                const _Float16* __restrict__ wPacked,
                _Float16* __restrict__ actOut,
                float* __restrict__ statSum,
                float* __restrict__ statSq,
                int HI, int WI, int HO, int WO, int py, int px)
{
    static_assert(CIN % 32 == 0 && COUT % 32 == 0, "tile divisibility");
    constexpr int NT  = COUT / 32;   // col subtiles per wave (WAVES_N=2)
    constexpr int NCT = COUT / 16;   // total 16-wide col tiles
    constexpr int CPT = CIN / 32;    // K chunks per tap
    constexpr int NCH = NTAPS * CPT; // total K chunks
    constexpr int LDA = 40;          // halves per LDS row (16B align + bank pad)

    __shared__ _Float16 As[2][128 * LDA];
    __shared__ float sSum[COUT];
    __shared__ float sSq[COUT];

    const int tid  = threadIdx.x;
    const int lane = tid & 31;
    const int wave = tid >> 5;
    const int wm   = wave >> 1;
    const int wn   = wave & 1;
    const int l16  = lane & 15;
    const bool hiL = lane >= 16;

    if (tid < COUT) { sSum[tid] = 0.f; sSq[tid] = 0.f; }

    const long Mtot  = (long)NIMG * HO * WO;
    const long mBase = (long)blockIdx.x * 128;

    // staging: each thread owns 16 halves (32B) of one A row per chunk
    const int sRow  = tid >> 1;
    const int sHalf = (tid & 1) * 16;
    const long gmS  = mBase + sRow;
    const bool rowValid = gmS < Mtot;
    int img = 0, oy = 0, ox = 0;
    if (rowValid) {
        img = (int)(gmS / (HO * WO));
        const int rem = (int)(gmS % (HO * WO));
        oy = rem / WO; ox = rem % WO;
    }

    auto stage = [&](int ch, int buf) {
        const int t   = ch / CPT;
        const int ci0 = (ch % CPT) * 32;
        int dy, dx;
        if (DECONV) {
            const int kyI = t >> 1, kxI = t & 1;
            dy = py ? (kyI ? 0 : 1) : (kyI ? -1 : 0);
            dx = px ? (kxI ? 0 : 1) : (kxI ? -1 : 0);
        } else {
            dy = t / 3 - 1; dx = t % 3 - 1;
        }
        const int iy = oy * S + dy;
        const int ix = ox * S + dx;
        const bool valid = rowValid && iy >= 0 && iy < HI && ix >= 0 && ix < WI;
        _Float16* dst = &As[buf][sRow * LDA + sHalf];
        if (valid) {
            const _Float16* src =
                actIn + (((long)img * HI + iy) * WI + ix) * CIN + (ci0 + sHalf);
            // low 32 bits of a flat shared pointer == wave-relative LDS offset
            const unsigned ldsOff = (unsigned)(size_t)dst;
            asm volatile("global_load_async_to_lds_b128 %0, %1, off"
                         :: "v"(ldsOff), "v"(src) : "memory");
            asm volatile("global_load_async_to_lds_b128 %0, %1, off offset:16"
                         :: "v"(ldsOff), "v"(src) : "memory");
        } else {
            const v8h zh = {};
            *(v8h*)(dst)     = zh;   // DScnt path; compiler waits before barrier
            *(v8h*)(dst + 8) = zh;
        }
    };

    v8f acc[2][NT];
#pragma unroll
    for (int a = 0; a < 2; ++a)
#pragma unroll
        for (int b = 0; b < NT; ++b) acc[a][b] = {};

    // prologue: fill buffer 0
    stage(0, 0);
    asm volatile("s_wait_asynccnt 0x0" ::: "memory");
    __syncthreads();

    for (int ch = 0; ch < NCH; ++ch) {
        const int cur = ch & 1;
        if (ch + 1 < NCH) {
            stage(ch + 1, cur ^ 1);   // async into the other buffer
            __builtin_prefetch(wPacked + ((long)(ch + 1) * NCT * 32 + lane) * 16, 0, 3);
        }

        // A fragments (contiguous k order, same as packed B): lane<16 K{0..15}, else K{16..31}
        v16h afr[2];
        const int kb = hiL ? 16 : 0;
#pragma unroll
        for (int mt = 0; mt < 2; ++mt) {
            const int row = wm * 32 + mt * 16 + l16;
            const v8h lo = *(const v8h*)&As[cur][row * LDA + kb];
            const v8h hp = *(const v8h*)&As[cur][row * LDA + 8 + kb];
            afr[mt] = __builtin_shufflevector(lo, hp,
                0,1,2,3,4,5,6,7,8,9,10,11,12,13,14,15);
        }
        // B fragments: contiguous 32B per lane from packed weights (L2 hot)
        v16h bfr[NT];
#pragma unroll
        for (int j = 0; j < NT; ++j) {
            const int ctG = wn * NT + j;
            const _Float16* bp = wPacked + (((long)ch * NCT + ctG) * 32 + lane) * 16;
            const v8h lo = *(const v8h*)bp;
            const v8h hp = *(const v8h*)(bp + 8);
            bfr[j] = __builtin_shufflevector(lo, hp,
                0,1,2,3,4,5,6,7,8,9,10,11,12,13,14,15);
        }
#pragma unroll
        for (int mt = 0; mt < 2; ++mt)
#pragma unroll
            for (int j = 0; j < NT; ++j)
                acc[mt][j] = __builtin_amdgcn_wmma_f32_16x16x32_f16(
                    false, afr[mt], false, bfr[j], (short)0, acc[mt][j], false, false);

        asm volatile("s_wait_asynccnt 0x0" ::: "memory"); // next-buffer fill done
        __syncthreads();                                  // + all reads of cur done
    }

    // Epilogue: raw fp16 NHWC store + BN partial sums.
    // C/D layout: VGPR g -> M = base + (lane>=16?8:0) + g ; N = lane%16
#pragma unroll
    for (int j = 0; j < NT; ++j) {
        const int co = wn * (NT * 16) + j * 16 + l16;
        float ps = 0.f, pq = 0.f;
#pragma unroll
        for (int mt = 0; mt < 2; ++mt) {
            const int rb = wm * 32 + mt * 16 + (hiL ? 8 : 0);
#pragma unroll
            for (int g = 0; g < 8; ++g) {
                const float v = acc[mt][j][g];
                ps += v; pq += v * v;   // padded rows are exact zeros -> harmless
                const long gm2 = mBase + rb + g;
                if (gm2 < Mtot) {
                    long pix;
                    if (DECONV) {
                        const int im  = (int)(gm2 / (HO * WO));
                        const int rem = (int)(gm2 % (HO * WO));
                        const int yy = rem / WO, xx = rem % WO;
                        pix = ((long)im * (2 * HO) + (2 * yy + py)) * (2 * WO)
                              + (2 * xx + px);
                    } else {
                        pix = gm2;
                    }
                    actOut[pix * COUT + co] = (_Float16)v;
                }
            }
        }
        ps += __shfl_xor(ps, 16, 32);
        pq += __shfl_xor(pq, 16, 32);
        if (!hiL) {
            atomicAdd(&sSum[co], ps);
            atomicAdd(&sSq[co], pq);
        }
    }
    __syncthreads();
    if (tid < COUT) {
        atomicAdd(&statSum[tid], sSum[tid]);
        atomicAdd(&statSq[tid], sSq[tid]);
    }
}

// ---------------------------------------------------------------------------
__global__ void bn_finalize(const float* __restrict__ sum, const float* __restrict__ sq,
                            const float* __restrict__ g, const float* __restrict__ b,
                            float invN, float* __restrict__ scale,
                            float* __restrict__ shift, int C)
{
    const int c = blockIdx.x * blockDim.x + threadIdx.x;
    if (c >= C) return;
    const float mu  = sum[c] * invN;
    const float var = sq[c] * invN - mu * mu;
    const float s   = g[c] * rsqrtf(var + 1e-5f);
    scale[c] = s;
    shift[c] = b[c] - mu * s;
}

// In-place BN affine + ReLU over raw fp16 NHWC activations (8 halves/thread).
__global__ void bn_apply(_Float16* __restrict__ x,
                         const float* __restrict__ scale,
                         const float* __restrict__ shift,
                         int C, long total8)
{
    const long i = (long)blockIdx.x * blockDim.x + threadIdx.x;
    if (i >= total8) return;
    const long base = i * 8;
    const int c0 = (int)(base % C);   // C multiple of 8 -> aligned channel run
    v8h v = *(v8h*)(x + base);
#pragma unroll
    for (int j = 0; j < 8; ++j)
        v[j] = (_Float16)fmaxf(0.f, (float)v[j] * scale[c0 + j] + shift[c0 + j]);
    *(v8h*)(x + base) = v;
}

__global__ void zero_f32(float* __restrict__ p, int n)
{
    const int i = blockIdx.x * blockDim.x + threadIdx.x;
    if (i < n) p[i] = 0.f;
}

// conv6: 3x3 32->1 over pre-normalized X5 -> cost[81][64*96] f32
__global__ void conv6_cost(const _Float16* __restrict__ x5,
                           const float* __restrict__ w6,
                           const float* __restrict__ b6,
                           float* __restrict__ cost)
{
    const int NPIX = IMGH * IMGW;
    const int idx = blockIdx.x * blockDim.x + threadIdx.x;
    if (idx >= NIMG * NPIX) return;
    const int img = idx / NPIX;
    const int pix = idx % NPIX;
    const int y = pix / IMGW, x = pix % IMGW;
    float acc = b6[0];
#pragma unroll
    for (int ky = 0; ky < 3; ++ky) {
        const int iy = y + ky - 1;
        if (iy < 0 || iy >= IMGH) continue;
#pragma unroll
        for (int kx = 0; kx < 3; ++kx) {
            const int ix = x + kx - 1;
            if (ix < 0 || ix >= IMGW) continue;
            const _Float16* p = x5 + (((long)img * IMGH + iy) * IMGW + ix) * 32;
#pragma unroll
            for (int c = 0; c < 32; ++c)
                acc += (float)p[c] * w6[(c * 3 + ky) * 3 + kx];
        }
    }
    cost[idx] = acc;
}

// softmax over 81 hypotheses + displacement expectation; out = coords + flow
__global__ void softmax_flow(const float* __restrict__ cost,
                             const float* __restrict__ coords,
                             float* __restrict__ out)
{
    const int NPIX = IMGH * IMGW;
    const int pix = blockIdx.x * blockDim.x + threadIdx.x;
    if (pix >= NPIX) return;
    float mx = -3.4e38f;
    for (int k = 0; k < NIMG; ++k) mx = fmaxf(mx, cost[(long)k * NPIX + pix]);
    float se = 0.f, su = 0.f, sv = 0.f;
    for (int k = 0; k < NIMG; ++k) {
        const float e = __expf(cost[(long)k * NPIX + pix] - mx);
        se += e;
        su += e * (float)(k % 9 - 4);
        sv += e * (float)(k / 9 - 4);
    }
    const float inv = 1.f / se;
    out[pix]        = coords[pix]        + su * inv;
    out[NPIX + pix] = coords[NPIX + pix] + sv * inv;
}

// ---------------------------------------------------------------------------
extern "C" void kernel_launch(void* const* d_in, const int* in_sizes, int n_in,
                              void* d_out, int out_size, void* d_ws, size_t ws_size,
                              hipStream_t stream)
{
    const float* feat1  = (const float*)d_in[0];
    const float* feat2  = (const float*)d_in[1];
    const float* coords = (const float*)d_in[2];
    const float* w1 = (const float*)d_in[3];
    const float* g1 = (const float*)d_in[4];
    const float* b1 = (const float*)d_in[5];
    const float* w2 = (const float*)d_in[6];
    const float* g2 = (const float*)d_in[7];
    const float* b2 = (const float*)d_in[8];
    const float* w3 = (const float*)d_in[9];
    const float* g3 = (const float*)d_in[10];
    const float* b3 = (const float*)d_in[11];
    const float* w4 = (const float*)d_in[12];
    const float* g4 = (const float*)d_in[13];
    const float* b4 = (const float*)d_in[14];
    const float* w5t = (const float*)d_in[15];
    const float* g5  = (const float*)d_in[16];
    const float* b5  = (const float*)d_in[17];
    const float* w6  = (const float*)d_in[18];
    const float* b6  = (const float*)d_in[19];
    float* out = (float*)d_out;

    char* ws = (char*)d_ws;
    size_t off = 0;
    auto alloc = [&](size_t bytes) {
        size_t o = off;
        off = (off + bytes + 255) & ~(size_t)255;
        return o;
    };

    const long NPIX_HI = (long)NIMG * 64 * 96;  // 497664
    const long NPIX_LO = (long)NIMG * 32 * 48;  // 124416

    // Region aliasing: R0 holds X0 / X2 / X4 ; R1 holds X1 / X3 / X5
    _Float16* R0 = (_Float16*)(ws + alloc((size_t)NPIX_HI * 64 * sizeof(_Float16)));
    _Float16* R1 = (_Float16*)(ws + alloc((size_t)NPIX_HI * 96 * sizeof(_Float16)));
    float* cost  = (float*)(ws + alloc((size_t)NPIX_HI * sizeof(float)));
    _Float16* pw1 = (_Float16*)(ws + alloc((size_t)18 * 6 * 512 * 2));
    _Float16* pw2 = (_Float16*)(ws + alloc((size_t)27 * 8 * 512 * 2));
    _Float16* pw3 = (_Float16*)(ws + alloc((size_t)36 * 8 * 512 * 2));
    _Float16* pw4 = (_Float16*)(ws + alloc((size_t)36 * 4 * 512 * 2));
    _Float16* pw5[4];
    for (int p = 0; p < 4; ++p)
        pw5[p] = (_Float16*)(ws + alloc((size_t)8 * 2 * 512 * 2));
    float* stats = (float*)(ws + alloc(2048 * sizeof(float)));

    float* sum1 = stats;       float* sq1 = sum1 + 96;
    float* sum2 = sq1 + 96;    float* sq2 = sum2 + 128;
    float* sum3 = sq2 + 128;   float* sq3 = sum3 + 128;
    float* sum4 = sq3 + 128;   float* sq4 = sum4 + 64;
    float* sum5 = sq4 + 64;    float* sq5 = sum5 + 32;
    float* sc1 = sq5 + 32;     float* sh1 = sc1 + 96;
    float* sc2 = sh1 + 96;     float* sh2 = sc2 + 128;
    float* sc3 = sh2 + 128;    float* sh3 = sc3 + 128;
    float* sc4 = sh3 + 128;    float* sh4 = sc4 + 64;
    float* sc5 = sh4 + 64;     float* sh5 = sc5 + 32;

    zero_f32<<<4, 256, 0, stream>>>(stats, 896);

    pack_conv_w<<<(18 * 6 * 512 + 255) / 256, 256, 0, stream>>>(w1, pw1, 64, 96, 3, 3);
    pack_conv_w<<<(27 * 8 * 512 + 255) / 256, 256, 0, stream>>>(w2, pw2, 96, 128, 3, 3);
    pack_conv_w<<<(36 * 8 * 512 + 255) / 256, 256, 0, stream>>>(w3, pw3, 128, 128, 3, 3);
    pack_conv_w<<<(36 * 4 * 512 + 255) / 256, 256, 0, stream>>>(w4, pw4, 128, 64, 3, 3);
    for (int p = 0; p < 4; ++p)
        pack_deconv_w<<<(8 * 2 * 512 + 255) / 256, 256, 0, stream>>>(w5t, pw5[p], p >> 1, p & 1);

    warp_concat_kernel<<<(int)((NPIX_HI + 255) / 256), 256, 0, stream>>>(feat1, feat2, coords, R0);

    const int blkHI = (int)((NPIX_HI + 127) / 128);
    const int blkLO = (int)((NPIX_LO + 127) / 128);
    auto blk8 = [](long total8) { return (int)((total8 + 255) / 256); };

    // L1: 64 -> 96, 64x96
    conv_igemm<64, 96, 1, 9, false><<<blkHI, 256, 0, stream>>>(
        R0, pw1, R1, sum1, sq1, 64, 96, 64, 96, 0, 0);
    bn_finalize<<<1, 96, 0, stream>>>(sum1, sq1, g1, b1, 1.f / (float)NPIX_HI, sc1, sh1, 96);
    bn_apply<<<blk8(NPIX_HI * 96 / 8), 256, 0, stream>>>(R1, sc1, sh1, 96, NPIX_HI * 96 / 8);

    // L2: 96 -> 128, s2 -> 32x48
    conv_igemm<96, 128, 2, 9, false><<<blkLO, 256, 0, stream>>>(
        R1, pw2, R0, sum2, sq2, 64, 96, 32, 48, 0, 0);
    bn_finalize<<<1, 128, 0, stream>>>(sum2, sq2, g2, b2, 1.f / (float)NPIX_LO, sc2, sh2, 128);
    bn_apply<<<blk8(NPIX_LO * 128 / 8), 256, 0, stream>>>(R0, sc2, sh2, 128, NPIX_LO * 128 / 8);

    // L3: 128 -> 128
    conv_igemm<128, 128, 1, 9, false><<<blkLO, 256, 0, stream>>>(
        R0, pw3, R1, sum3, sq3, 32, 48, 32, 48, 0, 0);
    bn_finalize<<<1, 128, 0, stream>>>(sum3, sq3, g3, b3, 1.f / (float)NPIX_LO, sc3, sh3, 128);
    bn_apply<<<blk8(NPIX_LO * 128 / 8), 256, 0, stream>>>(R1, sc3, sh3, 128, NPIX_LO * 128 / 8);

    // L4: 128 -> 64
    conv_igemm<128, 64, 1, 9, false><<<blkLO, 256, 0, stream>>>(
        R1, pw4, R0, sum4, sq4, 32, 48, 32, 48, 0, 0);
    bn_finalize<<<1, 64, 0, stream>>>(sum4, sq4, g4, b4, 1.f / (float)NPIX_LO, sc4, sh4, 64);
    bn_apply<<<blk8(NPIX_LO * 64 / 8), 256, 0, stream>>>(R0, sc4, sh4, 64, NPIX_LO * 64 / 8);

    // L5: deconv 64 -> 32, four output parities -> 64x96
    for (int p = 0; p < 4; ++p)
        conv_igemm<64, 32, 1, 4, true><<<blkLO, 256, 0, stream>>>(
            R0, pw5[p], R1, sum5, sq5, 32, 48, 32, 48, p >> 1, p & 1);
    bn_finalize<<<1, 32, 0, stream>>>(sum5, sq5, g5, b5, 1.f / (float)NPIX_HI, sc5, sh5, 32);
    bn_apply<<<blk8(NPIX_HI * 32 / 8), 256, 0, stream>>>(R1, sc5, sh5, 32, NPIX_HI * 32 / 8);

    // L6 + softmax-expectation
    conv6_cost<<<(int)((NPIX_HI + 255) / 256), 256, 0, stream>>>(R1, w6, b6, cost);
    softmax_flow<<<(64 * 96 + 255) / 256, 256, 0, stream>>>(cost, coords, out);
}
